// vn_encoder_73521250172984
// MI455X (gfx1250) — matrix operand order repaired and checked by reading, not compiled, not run on hardware
//
#include <hip/hip_runtime.h>

#define VN_EPS   1e-6f
#define VN_BNEPS 1e-5f
#define VN_NEG   0.2f

static constexpr int  NB  = 4;
static constexpr int  NP  = 2048;
static constexpr int  KNN = 20;
static constexpr long NC  = (long)NP * 3 * KNN;   // 122880 edge columns / batch
static constexpr long NC5 = (long)NP * 3;         // 6144 point columns / batch

typedef __attribute__((ext_vector_type(16))) _Float16 v16h;
typedef __attribute__((ext_vector_type(8)))  _Float16 v8h;
typedef __attribute__((ext_vector_type(8)))  float    v8f;

__device__ __forceinline__ void vn_wait_async0() {
#if __has_builtin(__builtin_amdgcn_s_wait_asynccnt)
  __builtin_amdgcn_s_wait_asynccnt(0);
#else
  asm volatile("s_wait_asynccnt 0x0" ::: "memory");
#endif
}

// ---------------------------------------------------------------- WMMA helpers
// A operand (16x32 f16): lane l (l15 = row m), half = l>>4.
//   elements e=0..15 -> k = half*16 + e (contiguous 16 halves)
__device__ __forceinline__ v16h vn_load_a(const _Float16* row, int k0, int hlf) {
  v8h lo = *reinterpret_cast<const v8h*>(row + k0 + hlf * 16);
  v8h hi = *reinterpret_cast<const v8h*>(row + k0 + hlf * 16 + 8);
  v16h r;
#pragma unroll
  for (int e = 0; e < 8; ++e) { r[e] = lo[e]; r[e + 8] = hi[e]; }
  return r;
}
// B operand (32x16 f16) from K-contiguous rows: lane -> col n, k = half*16 + e.
__device__ __forceinline__ v16h vn_load_b(const _Float16* row, int k0, int hlf) {
  v8h lo = *reinterpret_cast<const v8h*>(row + k0 + hlf * 16);
  v8h hi = *reinterpret_cast<const v8h*>(row + k0 + hlf * 16 + 8);
  v16h r;
#pragma unroll
  for (int e = 0; e < 8; ++e) { r[e] = lo[e]; r[e + 8] = hi[e]; }
  return r;
}

// Generic GEMM: C[Mp x cols] = A[Mp x Kp](f16) * BT[cols x Kp](f16)^T, f32 accum.
// 8 waves / block; block stages 128 Bt rows into LDS via async-to-LDS copies,
// double-buffered, 32 K-halves per step. Requires colTiles % 8 == 0 (true at
// every call site), so all waves participate in barriers and EXEC is full.
__global__ __launch_bounds__(256)
void vn_gemm(const _Float16* __restrict__ A, const _Float16* __restrict__ BT,
             float* __restrict__ C, const float* __restrict__ bias,
             int Kp, int colTiles, long sBT, long sC, int ldc)
{
  (void)colTiles;
  __shared__ _Float16 sB[2][128 * 32];   // 2 x 8KB staging buffers
  const int tid  = threadIdx.x;
  const int lane = tid & 31;
  const int wid  = tid >> 5;
  const int tm = blockIdx.y;
  const int b  = blockIdx.z;
  const int hlf = lane >> 4, l15 = lane & 15;
  const _Float16* Arow  = A + (long)(tm * 16 + l15) * Kp;
  const _Float16* Bbase = BT + (long)b * sBT + ((long)blockIdx.x * 128) * Kp;

  // Stage 32 halves (64B) for each of 128 rows: 512 x 16B chunks, 2 per thread.
  auto stage = [&](int buf, int k0) {
#pragma unroll
    for (int q = 0; q < 2; ++q) {
      int ch  = tid + q * 256;
      int r   = ch >> 2;
      int sub = ch & 3;
      const _Float16* src = Bbase + (long)r * Kp + k0 + sub * 8;
      unsigned dst = (unsigned)(size_t)(&sB[buf][r * 32 + sub * 8]);
      asm volatile("global_load_async_to_lds_b128 %0, %1, off"
                   :: "v"(dst), "v"(src) : "memory");
    }
  };

  stage(0, 0);
  v8f acc = {};
  int buf = 0;
  for (int k0 = 0; k0 < Kp; k0 += 32) {
    vn_wait_async0();      // our stage of 'buf' landed in LDS
    __syncthreads();       // everyone's stage of 'buf' landed; prev reads done
    if (k0 + 32 < Kp) {
      stage(buf ^ 1, k0 + 32);
      __builtin_prefetch(Arow + k0 + 32, 0, 1);
    }
    v16h av = vn_load_a(Arow, k0, hlf);
    const _Float16* lb = &sB[buf][(wid * 16 + l15) * 32 + hlf * 16];
    v8h lo = *reinterpret_cast<const v8h*>(lb);
    v8h hi = *reinterpret_cast<const v8h*>(lb + 8);
    v16h bv;
#pragma unroll
    for (int e = 0; e < 8; ++e) { bv[e] = lo[e]; bv[e + 8] = hi[e]; }
    acc = __builtin_amdgcn_wmma_f32_16x16x32_f16(false, av, false, bv,
                                                 (short)0, acc, false, false);
    buf ^= 1;
  }
  const int tn = blockIdx.x * 8 + wid;
  float* Cp = C + (long)b * sC + (long)(tm * 16) * ldc + tn * 16 + l15;
#pragma unroll
  for (int r = 0; r < 8; ++r) {
    int mrow = r + hlf * 8;
    float v = acc[r];
    if (bias) v += bias[tm * 16 + mrow];
    Cp[(long)mrow * ldc] = v;
  }
}

// Gram/kNN distance: dist[i][j] = 2*<x_i,x_j> - |x_i|^2 - |x_j|^2
__global__ __launch_bounds__(256)
void vn_gram(const _Float16* __restrict__ xfT, const float* __restrict__ xx,
             float* __restrict__ dist, int Dp)
{
  const int lane = threadIdx.x & 31, wid = threadIdx.x >> 5;
  const int tn = blockIdx.x * 8 + wid;
  const int tm = blockIdx.y, b = blockIdx.z;
  const int hlf = lane >> 4, l15 = lane & 15;
  const _Float16* base = xfT + (long)b * NP * Dp;
  const _Float16* Arow = base + (long)(tm * 16 + l15) * Dp;
  const _Float16* Brow = base + (long)(tn * 16 + l15) * Dp;
  v8f acc = {};
  for (int k0 = 0; k0 < Dp; k0 += 32) {
    v16h av = vn_load_a(Arow, k0, hlf);
    v16h bv = vn_load_b(Brow, k0, hlf);
    acc = __builtin_amdgcn_wmma_f32_16x16x32_f16(false, av, false, bv,
                                                 (short)0, acc, false, false);
  }
  const float* xb = xx + b * NP;
  float* D = dist + (long)b * NP * NP;
#pragma unroll
  for (int r = 0; r < 8; ++r) {
    int i = tm * 16 + r + hlf * 8;
    int j = tn * 16 + l15;
    D[(long)i * NP + j] = 2.f * acc[r] - xb[i] - xb[j];
  }
}

// ------------------------------------------------------------------- packing
__global__ void vn_pack_xfT(const float* __restrict__ x, _Float16* __restrict__ xfT,
                            float* __restrict__ xx, int D, int Dp)
{
  int t = blockIdx.x * blockDim.x + threadIdx.x;
  if (t >= NB * NP) return;
  int n = t % NP, b = t / NP;
  _Float16* o = xfT + (long)t * Dp;
  float s = 0.f;
  for (int d = 0; d < D; ++d) {
    _Float16 hv = (_Float16)x[((long)b * D + d) * NP + n];
    o[d] = hv;
    float fv = (float)hv;
    s += fv * fv;
  }
  for (int d = D; d < Dp; ++d) o[d] = (_Float16)0.f;
  xx[t] = s;
}

__global__ void vn_topk(const float* __restrict__ dist, int* __restrict__ idx)
{
  int t = blockIdx.x * blockDim.x + threadIdx.x;
  if (t >= NB * NP) return;
  const float* row = dist + (long)t * NP;
  float bv[KNN]; int bi[KNN];
#pragma unroll
  for (int q = 0; q < KNN; ++q) { bv[q] = -3.4e38f; bi[q] = 0; }
  for (int j = 0; j < NP; ++j) {
    float d = row[j];
    if (d > bv[KNN - 1]) {
      int p = KNN - 1;
      while (p > 0 && bv[p - 1] < d) { bv[p] = bv[p - 1]; bi[p] = bi[p - 1]; --p; }
      bv[p] = d; bi[p] = j;
    }
  }
  for (int q = 0; q < KNN; ++q) idx[(long)t * KNN + q] = bi[q];
}

__global__ void vn_gather(const float* __restrict__ x, const int* __restrict__ idx,
                          _Float16* __restrict__ ET, int Cprev, int Cinp)
{
  long t = (long)blockIdx.x * blockDim.x + threadIdx.x;
  long total = (long)NB * NP * 3 * KNN;
  if (t >= total) return;
  int kk = (int)(t % KNN); long r = t / KNN;
  int v = (int)(r % 3); r /= 3;
  int n = (int)(r % NP); int b = (int)(r / NP);
  int j = idx[((long)b * NP + n) * KNN + kk];
  _Float16* o = ET + t * Cinp;    // t == b*NC + (n*3+v)*KNN + kk
  const float* xb = x + (long)b * Cprev * 3 * NP;
  for (int c = 0; c < Cprev; ++c) {
    float cn = xb[((long)c * 3 + v) * NP + n];
    float cj = xb[((long)c * 3 + v) * NP + j];
    o[c]         = (_Float16)(cj - cn);
    o[Cprev + c] = (_Float16)cn;
  }
  for (int c = 2 * Cprev; c < Cinp; ++c) o[c] = (_Float16)0.f;
}

__global__ void vn_pack_w(const float* __restrict__ W, const float* __restrict__ Wd,
                          _Float16* __restrict__ Wst, int Cout, int Coutd,
                          int Cin, int Cinp, int Mp)
{
  int t = blockIdx.x * blockDim.x + threadIdx.x;
  if (t >= Mp * Cinp) return;
  int row = t / Cinp, col = t % Cinp;
  float v = 0.f;
  if (col < Cin) {
    if (row < Cout) v = W[(long)row * Cin + col];
    else if (row < Cout + Coutd) v = Wd[(long)(row - Cout) * Cin + col];
  }
  Wst[t] = (_Float16)v;
}

// ------------------------------------------------------------------ VN BN/act
__global__ void vn_bn_stats(const float* __restrict__ lin, float* __restrict__ s1,
                            float* __restrict__ s2, long sLin, int ldl, int items)
{
  __shared__ float r1[256], r2[256];
  int c = blockIdx.x;
  float a1 = 0.f, a2 = 0.f;
  long total = (long)NB * items;
  for (long t = (long)blockIdx.y * blockDim.x + threadIdx.x; t < total;
       t += (long)gridDim.y * blockDim.x) {
    int b = (int)(t / items);
    long i3 = (t % items) * 3;
    const float* p = lin + (long)b * sLin + (long)c * ldl + i3;
    float nrm = sqrtf(p[0]*p[0] + p[1]*p[1] + p[2]*p[2]) + VN_EPS;
    a1 += nrm; a2 += nrm * nrm;
  }
  r1[threadIdx.x] = a1; r2[threadIdx.x] = a2;
  __syncthreads();
  for (int s = 128; s > 0; s >>= 1) {
    if ((int)threadIdx.x < s) {
      r1[threadIdx.x] += r1[threadIdx.x + s];
      r2[threadIdx.x] += r2[threadIdx.x + s];
    }
    __syncthreads();
  }
  if (threadIdx.x == 0) { atomicAdd(&s1[c], r1[0]); atomicAdd(&s2[c], r2[0]); }
}

__global__ void vn_bn_finalize(const float* s1, const float* s2, float* mean,
                               float* rsig, int Cout, float cnt)
{
  int c = blockIdx.x * blockDim.x + threadIdx.x;
  if (c >= Cout) return;
  float m = s1[c] / cnt;
  float v = s2[c] / cnt - m * m;
  mean[c] = m;
  rsig[c] = rsqrtf(v + VN_BNEPS);
}

__global__ void vn_bn_apply(float* __restrict__ lin, const float* __restrict__ mean,
                            const float* __restrict__ rsig, const float* __restrict__ gamma,
                            const float* __restrict__ beta, int Cout, int dshare,
                            long sLin, int ldl, int items)
{
  long t = (long)blockIdx.x * blockDim.x + threadIdx.x;
  long total = (long)NB * Cout * items;
  if (t >= total) return;
  int it = (int)(t % items); long r = t / items;
  int c = (int)(r % Cout); int b = (int)(r / Cout);
  long i3 = (long)it * 3;
  float* p = lin + (long)b * sLin + (long)c * ldl + i3;
  int drow = dshare ? Cout : (Cout + c);
  const float* d = lin + (long)b * sLin + (long)drow * ldl + i3;
  float p0 = p[0], p1 = p[1], p2 = p[2];
  float nrm = sqrtf(p0*p0 + p1*p1 + p2*p2) + VN_EPS;
  float nb = (nrm - mean[c]) * rsig[c] * gamma[c] + beta[c];
  float sc = nb / nrm;
  p0 *= sc; p1 *= sc; p2 *= sc;
  float d0 = d[0], d1 = d[1], d2 = d[2];
  float dot = p0*d0 + p1*d1 + p2*d2;
  float dd  = d0*d0 + d1*d1 + d2*d2 + VN_EPS;
  float tq = dot / dd;
  float q0, q1, q2;
  if (dot >= 0.f) { q0 = p0; q1 = p1; q2 = p2; }
  else { q0 = p0 - tq*d0; q1 = p1 - tq*d1; q2 = p2 - tq*d2; }
  p[0] = VN_NEG*p0 + (1.f - VN_NEG)*q0;
  p[1] = VN_NEG*p1 + (1.f - VN_NEG)*q1;
  p[2] = VN_NEG*p2 + (1.f - VN_NEG)*q2;
}

// VN max pool over K neighbors: pick kk maximizing <h, Wp h>
__global__ void vn_pool(const float* __restrict__ lin, const float* __restrict__ Wp,
                        float* __restrict__ xout, int Cout, long sLin, int ldl)
{
  extern __shared__ float sh[];                 // [Cout][3][KNN]
  int n = blockIdx.x, b = blockIdx.y;
  const float* L = lin + (long)b * sLin;
  long colBase = (long)n * 3 * KNN;
  int tot = Cout * 3 * KNN;
  for (int t = threadIdx.x; t < tot; t += blockDim.x) {
    int c = t / (3 * KNN); int rem = t % (3 * KNN);
    sh[t] = L[(long)c * ldl + colBase + rem];
  }
  __syncthreads();
  for (int c = threadIdx.x; c < Cout; c += blockDim.x) {
    const float* wr = Wp + (long)c * Cout;
    float best = -3.4e38f; int bk = 0;
    for (int kk = 0; kk < KNN; ++kk) {
      float dot = 0.f;
      for (int v = 0; v < 3; ++v) {
        float dp = 0.f;
        const float* col = sh + v * KNN + kk;
        for (int c2 = 0; c2 < Cout; ++c2) dp += wr[c2] * col[c2 * 3 * KNN];
        dot += sh[c * 3 * KNN + v * KNN + kk] * dp;
      }
      if (dot > best) { best = dot; bk = kk; }
    }
    for (int v = 0; v < 3; ++v)
      xout[(((long)b * Cout + c) * 3 + v) * NP + n] = sh[c * 3 * KNN + v * KNN + bk];
  }
}

// --------------------------------------------------------------- later stages
__global__ void vn_pack_cat5(const float* __restrict__ x1, const float* __restrict__ x2,
                             const float* __restrict__ x3, const float* __restrict__ x4,
                             _Float16* __restrict__ ET)
{
  int t = blockIdx.x * blockDim.x + threadIdx.x;
  if (t >= NB * (int)NC5) return;
  int col = t % (int)NC5, b = t / (int)NC5;
  int n = col / 3, v = col % 3;
  _Float16* o = ET + (long)t * 192;
  int c = 0;
  for (int q = 0; q < 21; ++q) o[c++] = (_Float16)x1[(((long)b*21+q)*3+v)*NP+n];
  for (int q = 0; q < 21; ++q) o[c++] = (_Float16)x2[(((long)b*21+q)*3+v)*NP+n];
  for (int q = 0; q < 42; ++q) o[c++] = (_Float16)x3[(((long)b*42+q)*3+v)*NP+n];
  for (int q = 0; q < 85; ++q) o[c++] = (_Float16)x4[(((long)b*85+q)*3+v)*NP+n];
  for (; c < 192; ++c) o[c] = (_Float16)0.f;
}

__global__ void vn_hmean(const float* __restrict__ lin5, float* __restrict__ hmean,
                         long sLin, int ldl)
{
  __shared__ float rr[256];
  int cv = blockIdx.x, b = blockIdx.y;
  int c = cv / 3, v = cv % 3;
  float a = 0.f;
  for (int n = threadIdx.x; n < NP; n += 256)
    a += lin5[(long)b * sLin + (long)c * ldl + (long)n * 3 + v];
  rr[threadIdx.x] = a; __syncthreads();
  for (int s = 128; s > 0; s >>= 1) {
    if ((int)threadIdx.x < s) rr[threadIdx.x] += rr[threadIdx.x + s];
    __syncthreads();
  }
  if (threadIdx.x == 0) hmean[((long)b * 341 + c) * 3 + v] = rr[0] / NP;
}

__global__ void vn_pack_std1(const float* __restrict__ lin5, const float* __restrict__ hmean,
                             _Float16* __restrict__ ET, long sLin, int ldl)
{
  int t = blockIdx.x * blockDim.x + threadIdx.x;
  if (t >= NB * (int)NC5) return;
  int col = t % (int)NC5, b = t / (int)NC5;
  int v = col % 3;
  _Float16* o = ET + (long)t * 704;
  const float* L = lin5 + (long)b * sLin;
  for (int c = 0; c < 341; ++c) o[c]       = (_Float16)L[(long)c * ldl + col];
  for (int c = 0; c < 341; ++c) o[341 + c] = (_Float16)hmean[((long)b*341+c)*3+v];
  for (int c = 682; c < 704; ++c) o[c] = (_Float16)0.f;
}

__global__ void vn_pack_std2(const float* __restrict__ lin, _Float16* __restrict__ ET,
                             long sLin, int ldl)
{
  int t = blockIdx.x * blockDim.x + threadIdx.x;
  if (t >= NB * (int)NC5) return;
  int col = t % (int)NC5, b = t / (int)NC5;
  _Float16* o = ET + (long)t * 352;
  const float* L = lin + (long)b * sLin;
  for (int c = 0; c < 341; ++c) o[c] = (_Float16)L[(long)c * ldl + col];
  for (int c = 341; c < 352; ++c) o[c] = (_Float16)0.f;
}

__global__ void vn_stdlin(const float* __restrict__ lin2, const float* __restrict__ W,
                          float* __restrict__ z3, long sLin, int ldl)
{
  int t = blockIdx.x * blockDim.x + threadIdx.x;
  if (t >= NB * 9 * NP) return;
  int n = t % NP; int r = t / NP;
  int v = r % 3; r /= 3;
  int o = r % 3; int b = r / 3;
  float a = 0.f;
  const float* L = lin2 + (long)b * sLin;
  for (int c = 0; c < 170; ++c) a += W[o * 170 + c] * L[(long)c * ldl + (long)n * 3 + v];
  z3[(((long)b * 3 + o) * 3 + v) * NP + n] = a;
}

__global__ void vn_project(const float* __restrict__ lin5, const float* __restrict__ hmean,
                           const float* __restrict__ z3, float* __restrict__ hflat,
                           _Float16* __restrict__ hT, long sLin5, int ldl)
{
  long t = (long)blockIdx.x * blockDim.x + threadIdx.x;
  long total = (long)NB * 682 * NP;
  if (t >= total) return;
  int n = (int)(t % NP); long r = t / NP;
  int i = (int)(r % 682); int b = (int)(r / 682);
  float h0, h1, h2;
  if (i < 341) {
    const float* p = lin5 + (long)b * sLin5 + (long)i * ldl + (long)n * 3;
    h0 = p[0]; h1 = p[1]; h2 = p[2];
  } else {
    const float* p = hmean + ((long)b * 341 + (i - 341)) * 3;
    h0 = p[0]; h1 = p[1]; h2 = p[2];
  }
  const float* zb = z3 + (long)b * 9 * NP;
#pragma unroll
  for (int kd = 0; kd < 3; ++kd) {
    float val = h0 * zb[(0*3+kd)*NP + n] + h1 * zb[(1*3+kd)*NP + n] + h2 * zb[(2*3+kd)*NP + n];
    hflat[((long)b * 2046 + i * 3 + kd) * NP + n] = val;
    hT[((long)b * NP + n) * 2048 + i * 3 + kd] = (_Float16)val;
  }
}

__global__ void vn_gpool(const float* __restrict__ hflat, float* __restrict__ gmax,
                         float* __restrict__ gmean)
{
  __shared__ float rm[256], rs[256];
  int ch = blockIdx.x, b = blockIdx.y;
  const float* p = hflat + ((long)b * 2046 + ch) * NP;
  float mx = -3.4e38f, sm = 0.f;
  for (int n = threadIdx.x; n < NP; n += 256) { float v = p[n]; mx = fmaxf(mx, v); sm += v; }
  rm[threadIdx.x] = mx; rs[threadIdx.x] = sm; __syncthreads();
  for (int s = 128; s > 0; s >>= 1) {
    if ((int)threadIdx.x < s) {
      rm[threadIdx.x] = fmaxf(rm[threadIdx.x], rm[threadIdx.x + s]);
      rs[threadIdx.x] += rs[threadIdx.x + s];
    }
    __syncthreads();
  }
  if (threadIdx.x == 0) { gmax[b*2046+ch] = rm[0]; gmean[b*2046+ch] = rs[0] / NP; }
}

__global__ void vn_fc1(const float* __restrict__ gmax, const float* __restrict__ gmean,
                       const float* __restrict__ W, const float* __restrict__ bias,
                       float* __restrict__ g1)
{
  int t = blockIdx.x * blockDim.x + threadIdx.x;
  if (t >= NB * 512) return;
  int o = t % 512, b = t / 512;
  const float* w = W + (long)o * 4092;
  const float* gm = gmax + (long)b * 2046;
  const float* ga = gmean + (long)b * 2046;
  float a = bias[o];
  for (int c = 0; c < 2046; ++c) a += gm[c] * w[c];
  for (int c = 0; c < 2046; ++c) a += ga[c] * w[2046 + c];
  g1[t] = a;
}

__global__ void vn_fcbn(float* __restrict__ g1, const float* __restrict__ gg,
                        const float* __restrict__ gb)
{
  int o = blockIdx.x * blockDim.x + threadIdx.x;
  if (o >= 512) return;
  float v[NB]; float m = 0.f;
  for (int b = 0; b < NB; ++b) { v[b] = g1[b * 512 + o]; m += v[b]; }
  m /= NB;
  float var = 0.f;
  for (int b = 0; b < NB; ++b) { float d = v[b] - m; var += d * d; }
  var /= NB;
  float rs = rsqrtf(var + VN_BNEPS);
  for (int b = 0; b < NB; ++b) {
    float xv = (v[b] - m) * rs * gg[o] + gb[o];
    g1[b * 512 + o] = xv >= 0.f ? xv : VN_NEG * xv;
  }
}

__global__ void vn_fc2(const float* __restrict__ g1, const float* __restrict__ W,
                       const float* __restrict__ bias, float* __restrict__ out)
{
  int t = blockIdx.x * blockDim.x + threadIdx.x;
  if (t >= NB * 256) return;
  int o = t % 256, b = t / 256;
  float a = bias[o];
  for (int c = 0; c < 512; ++c) a += g1[b * 512 + c] * W[o * 512 + c];
  out[t] = a;
}

// --------------------------------------------------------------- ws layout
static constexpr size_t O_X1    = 0;
static constexpr size_t O_X2    = O_X1   + (size_t)NB*21*3*NP*4;
static constexpr size_t O_X3    = O_X2   + (size_t)NB*21*3*NP*4;
static constexpr size_t O_X4    = O_X3   + (size_t)NB*42*3*NP*4;
static constexpr size_t O_XFT   = O_X4   + (size_t)NB*85*3*NP*4;
static constexpr size_t O_XX    = O_XFT  + (size_t)NB*NP*128*2;
static constexpr size_t O_IDX   = O_XX   + (size_t)NB*NP*4;
static constexpr size_t O_WST   = O_IDX  + (size_t)NB*NP*KNN*4;
static constexpr size_t O_S1    = O_WST  + 1048576;
static constexpr size_t O_S2    = O_S1   + 4096;
static constexpr size_t O_MEAN  = O_S2   + 4096;
static constexpr size_t O_RSIG  = O_MEAN + 4096;
static constexpr size_t O_HMEAN = O_RSIG + 4096;
static constexpr size_t O_GMAX  = O_HMEAN+ 16384;
static constexpr size_t O_GMEAN = O_GMAX + 32768;
static constexpr size_t O_G1    = O_GMEAN+ 32768;
static constexpr size_t O_Z3    = O_G1   + 8192;
static constexpr size_t O_R2    = O_Z3   + (size_t)NB*9*NP*4;      // f16 staging (94 MB)
static constexpr size_t O_R1    = O_R2   + (size_t)NB*NC*96*2;     // dist / lin arena (346 MB)
static constexpr size_t O_LIN   = O_R1;
static constexpr size_t O_LIN5  = O_R1;
static constexpr size_t O_LS1   = O_LIN5 + (size_t)NB*352*NC5*4;
static constexpr size_t O_LS2   = O_LS1  + (size_t)NB*688*NC5*4;
static constexpr size_t O_HFL   = O_LS2  + (size_t)NB*352*NC5*4;

extern "C" void kernel_launch(void* const* d_in, const int* in_sizes, int n_in,
                              void* d_out, int out_size, void* d_ws, size_t ws_size,
                              hipStream_t stream) {
  (void)in_sizes; (void)n_in; (void)out_size; (void)ws_size;
  char* ws = (char*)d_ws;
  auto WF = [&](size_t o){ return (float*)(ws + o); };
  auto WH = [&](size_t o){ return (_Float16*)(ws + o); };
  auto WI = [&](size_t o){ return (int*)(ws + o); };
  auto IN = [&](int i){ return (const float*)d_in[i]; };

  struct ECfg { int Cprev, D, Dp, Cinp, Cout, Mp, wI, poolI; };
  const ECfg L[4] = {
    { 1,   3,  32, 32, 21,  48,  1, 30},
    {21,  63,  64, 64, 21,  48,  5, 31},
    {21,  63,  64, 64, 42,  96,  9, 32},
    {42, 126, 128, 96, 85, 176, 13, 33},
  };
  float* xs[4] = { WF(O_X1), WF(O_X2), WF(O_X3), WF(O_X4) };
  const float* xin = IN(0);

  // -------- edge-conv blocks 1..4
  for (int l = 0; l < 4; ++l) {
    const ECfg& c = L[l];
    vn_pack_xfT<<<32, 256, 0, stream>>>(xin, WH(O_XFT), WF(O_XX), c.D, c.Dp);
    vn_gram<<<dim3(16, 128, NB), 256, 0, stream>>>(WH(O_XFT), WF(O_XX), WF(O_LIN), c.Dp);
    vn_topk<<<32, 256, 0, stream>>>(WF(O_LIN), WI(O_IDX));
    long gt = (long)NB * NP * 3 * KNN;
    vn_gather<<<(int)((gt + 255) / 256), 256, 0, stream>>>(xin, WI(O_IDX), WH(O_R2),
                                                           c.Cprev, c.Cinp);
    vn_pack_w<<<(c.Mp * c.Cinp + 255) / 256, 256, 0, stream>>>(IN(c.wI), IN(c.wI + 1),
        WH(O_WST), c.Cout, c.Cout, 2 * c.Cprev, c.Cinp, c.Mp);
    vn_gemm<<<dim3(960, c.Mp / 16, NB), 256, 0, stream>>>(WH(O_WST), WH(O_R2), WF(O_LIN),
        nullptr, c.Cinp, (int)(NC / 16), NC * c.Cinp, (long)c.Mp * NC, (int)NC);
    hipMemsetAsync(ws + O_S1, 0, 4096, stream);
    hipMemsetAsync(ws + O_S2, 0, 4096, stream);
    vn_bn_stats<<<dim3(c.Cout, 32), 256, 0, stream>>>(WF(O_LIN), WF(O_S1), WF(O_S2),
        (long)c.Mp * NC, (int)NC, (int)(NC / 3));
    vn_bn_finalize<<<(c.Cout + 255) / 256, 256, 0, stream>>>(WF(O_S1), WF(O_S2),
        WF(O_MEAN), WF(O_RSIG), c.Cout, (float)((long)NB * NP * KNN));
    long at = (long)NB * c.Cout * (NC / 3);
    vn_bn_apply<<<(int)((at + 255) / 256), 256, 0, stream>>>(WF(O_LIN), WF(O_MEAN),
        WF(O_RSIG), IN(c.wI + 2), IN(c.wI + 3), c.Cout, 0, (long)c.Mp * NC, (int)NC,
        (int)(NC / 3));
    vn_pool<<<dim3(NP, NB), 256, c.Cout * 3 * KNN * 4, stream>>>(WF(O_LIN), IN(c.poolI),
        xs[l], c.Cout, (long)c.Mp * NC, (int)NC);
    xin = xs[l];
  }

  // -------- conv5 (169 -> 341, shared direction)
  vn_pack_cat5<<<(NB * (int)NC5 + 255) / 256, 256, 0, stream>>>(xs[0], xs[1], xs[2], xs[3],
                                                               WH(O_R2));
  vn_pack_w<<<(352 * 192 + 255) / 256, 256, 0, stream>>>(IN(17), IN(18), WH(O_WST),
      341, 1, 169, 192, 352);
  vn_gemm<<<dim3(48, 22, NB), 256, 0, stream>>>(WH(O_WST), WH(O_R2), WF(O_LIN5), nullptr,
      192, (int)(NC5 / 16), NC5 * 192, (long)352 * NC5, (int)NC5);
  hipMemsetAsync(ws + O_S1, 0, 4096, stream);
  hipMemsetAsync(ws + O_S2, 0, 4096, stream);
  vn_bn_stats<<<dim3(341, 32), 256, 0, stream>>>(WF(O_LIN5), WF(O_S1), WF(O_S2),
      (long)352 * NC5, (int)NC5, (int)(NC5 / 3));
  vn_bn_finalize<<<2, 256, 0, stream>>>(WF(O_S1), WF(O_S2), WF(O_MEAN), WF(O_RSIG),
      341, (float)(NB * NP));
  {
    long at = (long)NB * 341 * (NC5 / 3);
    vn_bn_apply<<<(int)((at + 255) / 256), 256, 0, stream>>>(WF(O_LIN5), WF(O_MEAN),
        WF(O_RSIG), IN(19), IN(20), 341, 1, (long)352 * NC5, (int)NC5, (int)(NC5 / 3));
  }

  // -------- std feature block
  vn_hmean<<<dim3(1023, NB), 256, 0, stream>>>(WF(O_LIN5), WF(O_HMEAN), (long)352 * NC5, (int)NC5);
  vn_pack_std1<<<(NB * (int)NC5 + 255) / 256, 256, 0, stream>>>(WF(O_LIN5), WF(O_HMEAN),
      WH(O_R2), (long)352 * NC5, (int)NC5);
  vn_pack_w<<<(688 * 704 + 255) / 256, 256, 0, stream>>>(IN(21), IN(22), WH(O_WST),
      341, 341, 682, 704, 688);
  vn_gemm<<<dim3(48, 43, NB), 256, 0, stream>>>(WH(O_WST), WH(O_R2), WF(O_LS1), nullptr,
      704, (int)(NC5 / 16), NC5 * 704, (long)688 * NC5, (int)NC5);
  hipMemsetAsync(ws + O_S1, 0, 4096, stream);
  hipMemsetAsync(ws + O_S2, 0, 4096, stream);
  vn_bn_stats<<<dim3(341, 32), 256, 0, stream>>>(WF(O_LS1), WF(O_S1), WF(O_S2),
      (long)688 * NC5, (int)NC5, (int)(NC5 / 3));
  vn_bn_finalize<<<2, 256, 0, stream>>>(WF(O_S1), WF(O_S2), WF(O_MEAN), WF(O_RSIG),
      341, (float)(NB * NP));
  {
    long at = (long)NB * 341 * (NC5 / 3);
    vn_bn_apply<<<(int)((at + 255) / 256), 256, 0, stream>>>(WF(O_LS1), WF(O_MEAN),
        WF(O_RSIG), IN(23), IN(24), 341, 0, (long)688 * NC5, (int)NC5, (int)(NC5 / 3));
  }
  vn_pack_std2<<<(NB * (int)NC5 + 255) / 256, 256, 0, stream>>>(WF(O_LS1), WH(O_R2),
      (long)688 * NC5, (int)NC5);
  vn_pack_w<<<(352 * 352 + 255) / 256, 256, 0, stream>>>(IN(25), IN(26), WH(O_WST),
      170, 170, 341, 352, 352);
  vn_gemm<<<dim3(48, 22, NB), 256, 0, stream>>>(WH(O_WST), WH(O_R2), WF(O_LS2), nullptr,
      352, (int)(NC5 / 16), NC5 * 352, (long)352 * NC5, (int)NC5);
  hipMemsetAsync(ws + O_S1, 0, 4096, stream);
  hipMemsetAsync(ws + O_S2, 0, 4096, stream);
  vn_bn_stats<<<dim3(170, 32), 256, 0, stream>>>(WF(O_LS2), WF(O_S1), WF(O_S2),
      (long)352 * NC5, (int)NC5, (int)(NC5 / 3));
  vn_bn_finalize<<<1, 256, 0, stream>>>(WF(O_S1), WF(O_S2), WF(O_MEAN), WF(O_RSIG),
      170, (float)(NB * NP));
  {
    long at = (long)NB * 170 * (NC5 / 3);
    vn_bn_apply<<<(int)((at + 255) / 256), 256, 0, stream>>>(WF(O_LS2), WF(O_MEAN),
        WF(O_RSIG), IN(27), IN(28), 170, 0, (long)352 * NC5, (int)NC5, (int)(NC5 / 3));
  }
  vn_stdlin<<<(NB * 9 * NP + 255) / 256, 256, 0, stream>>>(WF(O_LS2), IN(29), WF(O_Z3),
      (long)352 * NC5, (int)NC5);

  // -------- invariant projection + heads
  hipMemsetAsync(ws + O_R2, 0, (size_t)NB * NP * 2048 * 2, stream);  // hT (with pad cols)
  {
    long pt = (long)NB * 682 * NP;
    vn_project<<<(int)((pt + 255) / 256), 256, 0, stream>>>(WF(O_LIN5), WF(O_HMEAN),
        WF(O_Z3), WF(O_HFL), WH(O_R2), (long)352 * NC5, (int)NC5);
  }
  vn_gpool<<<dim3(2046, NB), 256, 0, stream>>>(WF(O_HFL), WF(O_GMAX), WF(O_GMEAN));
  vn_fc1<<<(NB * 512 + 255) / 256, 256, 0, stream>>>(WF(O_GMAX), WF(O_GMEAN), IN(34),
      IN(35), WF(O_G1));
  vn_fcbn<<<2, 256, 0, stream>>>(WF(O_G1), IN(36), IN(37));
  vn_fc2<<<(NB * 256 + 255) / 256, 256, 0, stream>>>(WF(O_G1), IN(38), IN(39),
      (float*)d_out);
  // point_feat = pp_W x hflat + pp_b  (WMMA, writes straight into d_out)
  vn_pack_w<<<(256 * 2048 + 255) / 256, 256, 0, stream>>>(IN(40), nullptr, WH(O_WST),
      256, 0, 2046, 2048, 256);
  vn_gemm<<<dim3(16, 16, NB), 256, 0, stream>>>(WH(O_WST), WH(O_R2),
      (float*)d_out + 1024, IN(41), 2048, NP / 16, (long)NP * 2048, (long)256 * NP, NP);
}
